// GeometricGAT_34608846471340
// MI455X (gfx1250) — matrix operand order repaired and checked by reading, not compiled, not run on hardware
//
#include <hip/hip_runtime.h>

#define N_NODES 50000
#define E_EDGES 800000
#define HEADS   4
#define FDIM    64
#define CIN     256
#define XDIM    192
#define NEG_SLOPE 0.2f
#define LDS_STRIDE 264   // halves per row (528B) -> conflict-free across 64x4B banks

typedef __attribute__((ext_vector_type(16))) _Float16 v16h;
typedef __attribute__((ext_vector_type(8)))  _Float16 v8h;
typedef __attribute__((ext_vector_type(8)))  float    v8f;

// ---------- helpers ----------

// order-preserving float->uint encoding for atomicMax on floats
__device__ __forceinline__ unsigned fenc(float f) {
  unsigned u = __float_as_uint(f);
  return (u & 0x80000000u) ? ~u : (u | 0x80000000u);
}
__device__ __forceinline__ float fdec(unsigned u) {
  return (u & 0x80000000u) ? __uint_as_float(u & 0x7fffffffu) : __uint_as_float(~u);
}

// Stage a 16x256-half A panel (8KB, rows contiguous in global) into LDS with a
// padded row stride, using gfx1250 async global->LDS b128 copies (ASYNCcnt).
// Generic shared pointers on AMDGPU are {aperture_hi32, lds_offset_lo32}, so
// truncating to 32 bits yields the LDS byte offset the instruction wants.
__device__ __forceinline__ void stage_a_panel(const _Float16* __restrict__ gsrc,
                                              _Float16* lds_dst, int tid) {
  // 512 chunks of 16B; 128 threads x 4 chunks, coalesced
  #pragma unroll
  for (int i = 0; i < 4; ++i) {
    int c = i * 128 + tid;
    int row = c >> 5, within = c & 31;
    unsigned loff = (unsigned)(unsigned long long)(lds_dst + row * LDS_STRIDE + within * 8);
    unsigned long long ga = (unsigned long long)(gsrc + c * 8);
    asm volatile("global_load_async_to_lds_b128 %0, %1, off"
                 :: "v"(loff), "v"(ga) : "memory");
  }
  asm volatile("s_wait_asynccnt 0x0" ::: "memory");
  __syncthreads();
}

// A-matrix 16x32 f16 fragment from LDS panel (ISA 7.12.2 layout).
// lanes 0-15: M = lane, K = k0+{0..7} then k0+{16..23}
// lanes 16-31: M = lane-16, K = k0+{8..15} then k0+{24..31}
__device__ __forceinline__ v16h lds_load_a16x32(const _Float16* As, int k0, int lane) {
  int row  = lane & 15;
  int kofs = k0 + ((lane >> 4) << 3);
  const _Float16* p = As + row * LDS_STRIDE + kofs;
  v8h lo = *(const v8h*)(p);
  v8h hi = *(const v8h*)(p + 16);
  return __builtin_shufflevector(lo, hi, 0,1,2,3,4,5,6,7,8,9,10,11,12,13,14,15);
}

// B-matrix 32x16 f16 tile loader: B[k][n] = W[o_base+n][k]  (W row-major [out,in])
// lanes 0-15: n = lane, K = k0+{0..7} then k0+{16..23}; lanes 16-31: n = lane-16, K = k0+{8..15} then k0+{24..31}
__device__ __forceinline__ v16h load_b32x16(const _Float16* __restrict__ W,
                                            int o_base, int k0, int lane, int ldk) {
  int row  = o_base + (lane & 15);
  int kofs = k0 + ((lane >> 4) << 3);
  const _Float16* p = W + (size_t)row * ldk + kofs;
  v8h lo = *(const v8h*)(p);
  v8h hi = *(const v8h*)(p + 16);
  return __builtin_shufflevector(lo, hi, 0,1,2,3,4,5,6,7,8,9,10,11,12,13,14,15);
}

__device__ __forceinline__ float wave_sum32(float v) {
  #pragma unroll
  for (int o = 16; o >= 1; o >>= 1) v += __shfl_xor(v, o, 32);
  return v;
}

// ---------- kernels ----------

__global__ __launch_bounds__(256) void k_cvt_weights(
    const float* __restrict__ wl, const float* __restrict__ wr,
    const float* __restrict__ ws, const float* __restrict__ wp,
    _Float16* __restrict__ hl, _Float16* __restrict__ hr,
    _Float16* __restrict__ hs, _Float16* __restrict__ hp) {
  int i = blockIdx.x * 256 + threadIdx.x;
  if (i < CIN * CIN) {
    hl[i] = (_Float16)wl[i];
    hr[i] = (_Float16)wr[i];
    hs[i] = (_Float16)ws[i];
    hp[i] = (_Float16)wp[i];
  }
}

__global__ __launch_bounds__(256) void k_init(float* __restrict__ outagg,
                                              unsigned* __restrict__ nmax,
                                              float* __restrict__ nsum) {
  int i = blockIdx.x * 256 + threadIdx.x;
  if (i < N_NODES * CIN) outagg[i] = 0.0f;
  if (i < N_NODES * HEADS) { nmax[i] = fenc(-3.0e38f); nsum[i] = 0.0f; }
}

// positional encoder MLP + concat -> f16 x_comb. One wave32 per node.
__global__ __launch_bounds__(128) void k_posenc(
    const float* __restrict__ x, const float* __restrict__ kpts,
    const float* __restrict__ pts3d,
    const float* __restrict__ w1, const float* __restrict__ b1,
    const float* __restrict__ g1, const float* __restrict__ bn1,
    const float* __restrict__ w2, const float* __restrict__ b2,
    const float* __restrict__ g2, const float* __restrict__ bn2,
    _Float16* __restrict__ Xc) {
  int node = blockIdx.x * 4 + (threadIdx.x >> 5);
  int lane = threadIdx.x & 31;
  if (node >= N_NODES) return;

  float u = kpts[node * 2 + 0] * (1.0f / 1216.0f);
  float v = kpts[node * 2 + 1] * (1.0f / 352.0f);
  float d = pts3d[node * 3 + 2];
  d = fminf(fmaxf(d, 0.1f), 100.0f);

  // layer 1: hidden[lane] over 32 units
  float h1 = w1[lane * 3 + 0] * u + w1[lane * 3 + 1] * v + w1[lane * 3 + 2] * d + b1[lane];
  float mean = wave_sum32(h1) * (1.0f / 32.0f);
  float df = h1 - mean;
  float var = wave_sum32(df * df) * (1.0f / 32.0f);
  float a1 = df * rsqrtf(var + 1e-5f) * g1[lane] + bn1[lane];
  a1 = a1 / (1.0f + __expf(-a1));  // SiLU

  // layer 2: 64 outputs, 2 per lane (lane, lane+32)
  float o0 = b2[lane], o1 = b2[lane + 32];
  #pragma unroll 4
  for (int k = 0; k < 32; ++k) {
    float hk = __shfl(a1, k, 32);
    o0 += w2[lane * 32 + k] * hk;
    o1 += w2[(lane + 32) * 32 + k] * hk;
  }
  float m2 = wave_sum32(o0 + o1) * (1.0f / 64.0f);
  float d0 = o0 - m2, d1 = o1 - m2;
  float v2 = wave_sum32(d0 * d0 + d1 * d1) * (1.0f / 64.0f);
  float r2 = rsqrtf(v2 + 1e-5f);
  float p0 = d0 * r2 * g2[lane] + bn2[lane];
  float p1 = d1 * r2 * g2[lane + 32] + bn2[lane + 32];

  _Float16* row = Xc + (size_t)node * CIN;
  #pragma unroll
  for (int j = 0; j < 6; ++j) {
    int c = lane + 32 * j;  // 0..191
    row[c] = (_Float16)x[(size_t)node * XDIM + c];
  }
  row[XDIM + lane]      = (_Float16)p0;
  row[XDIM + 32 + lane] = (_Float16)p1;
}

// Y{l,r,s} = Xc @ W{l,r,s}^T + b.
// Block = 4 waves; A panel staged once in LDS (async), shared by all waves;
// each wave owns one 16-col output tile -> 3 WMMAs per K-step.
__global__ __launch_bounds__(128) void k_gemm3(
    const _Float16* __restrict__ X,
    const _Float16* __restrict__ Wl, const _Float16* __restrict__ Wr,
    const _Float16* __restrict__ Ws,
    const float* __restrict__ bl, const float* __restrict__ br,
    const float* __restrict__ bs,
    float* __restrict__ Yl, float* __restrict__ Yr, float* __restrict__ Ys) {
  __shared__ _Float16 As[16 * LDS_STRIDE];
  int tid    = threadIdx.x;
  int lane   = tid & 31;
  int wave   = tid >> 5;
  int m_base = blockIdx.x * 16;
  int o_base = (blockIdx.y * 4 + wave) * 16;

  stage_a_panel(X + (size_t)m_base * CIN, As, tid);

  v8f cl = {}, cr = {}, cs = {};
  #pragma unroll
  for (int k0 = 0; k0 < CIN; k0 += 32) {
    v16h a  = lds_load_a16x32(As, k0, lane);
    if (k0 + 32 < CIN)
      __builtin_prefetch(Wl + (size_t)(o_base + (lane & 15)) * CIN + k0 + 32, 0, 3);
    v16h tl = load_b32x16(Wl, o_base, k0, lane, CIN);
    v16h tr = load_b32x16(Wr, o_base, k0, lane, CIN);
    v16h ts = load_b32x16(Ws, o_base, k0, lane, CIN);
    cl = __builtin_amdgcn_wmma_f32_16x16x32_f16(false, a, false, tl, (short)0, cl, false, false);
    cr = __builtin_amdgcn_wmma_f32_16x16x32_f16(false, a, false, tr, (short)0, cr, false, false);
    cs = __builtin_amdgcn_wmma_f32_16x16x32_f16(false, a, false, ts, (short)0, cs, false, false);
  }

  int ncol = o_base + (lane & 15);
  int mrow = m_base + ((lane >> 4) << 3);
  float abl = bl[ncol], abr = br[ncol], abs_ = bs[ncol];
  #pragma unroll
  for (int r = 0; r < 8; ++r) {
    size_t idx = (size_t)(mrow + r) * CIN + ncol;
    Yl[idx] = cl[r] + abl;
    Yr[idx] = cr[r] + abr;
    Ys[idx] = cs[r] + abs_;
  }
}

// edge pass 1: edge_attr, logits, segment-max (encoded atomicMax)
__global__ __launch_bounds__(256) void k_edge_logits(
    const int* __restrict__ ei, const float* __restrict__ kpts,
    const float* __restrict__ xl, const float* __restrict__ xr,
    const float* __restrict__ lew, const float* __restrict__ att,
    float* __restrict__ logits, unsigned* __restrict__ nmax,
    float* __restrict__ out_ea) {
  int t = blockIdx.x * 256 + threadIdx.x;
  if (t >= E_EDGES * HEADS) return;
  int e = t >> 2, h = t & 3;
  int src = ei[e], dst = ei[E_EDGES + e];

  float r0 = (kpts[dst * 2 + 0] - kpts[src * 2 + 0]) * (1.0f / 1216.0f);
  float r1 = (kpts[dst * 2 + 1] - kpts[src * 2 + 1]) * (1.0f / 352.0f);
  float dd = sqrtf(r0 * r0 + r1 * r1);
  if (h == 0) {
    out_ea[(size_t)e * 3 + 0] = r0;
    out_ea[(size_t)e * 3 + 1] = r1;
    out_ea[(size_t)e * 3 + 2] = dd;
  }

  const float* pl = xl + (size_t)src * CIN + h * FDIM;
  const float* pr = xr + (size_t)dst * CIN + h * FDIM;
  const float* pw = lew + (size_t)(h * FDIM) * 3;
  const float* pa = att + h * FDIM;
  float logit = 0.0f;
  #pragma unroll 4
  for (int f = 0; f < FDIM; ++f) {
    float ee = pw[f * 3 + 0] * r0 + pw[f * 3 + 1] * r1 + pw[f * 3 + 2] * dd;
    float z = pl[f] + pr[f] + ee;
    z = (z > 0.0f) ? z : NEG_SLOPE * z;
    logit += z * pa[f];
  }
  logits[t] = logit;
  atomicMax(&nmax[dst * HEADS + h], fenc(logit));
}

// edge pass 2: exp(logit - max), segment-sum
__global__ __launch_bounds__(256) void k_edge_exp(
    const int* __restrict__ ei, float* __restrict__ logits,
    const unsigned* __restrict__ nmax, float* __restrict__ nsum) {
  int t = blockIdx.x * 256 + threadIdx.x;
  if (t >= E_EDGES * HEADS) return;
  int e = t >> 2, h = t & 3;
  int dst = ei[E_EDGES + e];
  float ex = __expf(logits[t] - fdec(nmax[dst * HEADS + h]));
  logits[t] = ex;
  atomicAdd(&nsum[dst * HEADS + h], ex);
}

// edge pass 3: alpha + weighted scatter-aggregate
__global__ __launch_bounds__(256) void k_edge_scatter(
    const int* __restrict__ ei, const float* __restrict__ ex,
    const float* __restrict__ nsum, const float* __restrict__ xl,
    float* __restrict__ outagg, float* __restrict__ out_alpha) {
  int t = blockIdx.x * 256 + threadIdx.x;
  if (t >= E_EDGES * HEADS) return;
  int e = t >> 2, h = t & 3;
  int src = ei[e], dst = ei[E_EDGES + e];
  float alpha = ex[t] / (nsum[dst * HEADS + h] + 1e-16f);
  out_alpha[t] = alpha;
  const float* pl = xl + (size_t)src * CIN + h * FDIM;
  float* po = outagg + (size_t)dst * CIN + h * FDIM;
  #pragma unroll 4
  for (int f = 0; f < FDIM; ++f) atomicAdd(&po[f], pl[f] * alpha);
}

// epilogue: +conv_b, LN(256), +identity, SiLU -> f16 act. One wave per node.
__global__ __launch_bounds__(128) void k_epilogue(
    const float* __restrict__ outagg, const float* __restrict__ conv_b,
    const float* __restrict__ ng, const float* __restrict__ nb,
    const float* __restrict__ ident, _Float16* __restrict__ act) {
  int node = blockIdx.x * 4 + (threadIdx.x >> 5);
  int lane = threadIdx.x & 31;
  if (node >= N_NODES) return;

  float vals[8];
  float psum = 0.0f;
  #pragma unroll
  for (int i = 0; i < 8; ++i) {
    int c = lane * 8 + i;
    vals[i] = outagg[(size_t)node * CIN + c] + conv_b[c];
    psum += vals[i];
  }
  float mean = wave_sum32(psum) * (1.0f / 256.0f);
  float pvar = 0.0f;
  #pragma unroll
  for (int i = 0; i < 8; ++i) { float dv = vals[i] - mean; pvar += dv * dv; }
  float rstd = rsqrtf(wave_sum32(pvar) * (1.0f / 256.0f) + 1e-5f);
  #pragma unroll
  for (int i = 0; i < 8; ++i) {
    int c = lane * 8 + i;
    float v = (vals[i] - mean) * rstd * ng[c] + nb[c];
    v += ident[(size_t)node * CIN + c];
    v = v / (1.0f + __expf(-v));  // SiLU
    act[(size_t)node * CIN + c] = (_Float16)v;
  }
}

// final projection GEMM: out = act @ proj_w^T + proj_b (LDS-staged A panel)
__global__ __launch_bounds__(128) void k_proj(
    const _Float16* __restrict__ X, const _Float16* __restrict__ W,
    const float* __restrict__ b, float* __restrict__ Y) {
  __shared__ _Float16 As[16 * LDS_STRIDE];
  int tid    = threadIdx.x;
  int lane   = tid & 31;
  int wave   = tid >> 5;
  int m_base = blockIdx.x * 16;
  int o_base = (blockIdx.y * 4 + wave) * 16;

  stage_a_panel(X + (size_t)m_base * CIN, As, tid);

  v8f c = {};
  #pragma unroll
  for (int k0 = 0; k0 < CIN; k0 += 32) {
    v16h a  = lds_load_a16x32(As, k0, lane);
    if (k0 + 32 < CIN)
      __builtin_prefetch(W + (size_t)(o_base + (lane & 15)) * CIN + k0 + 32, 0, 3);
    v16h tb = load_b32x16(W, o_base, k0, lane, CIN);
    c = __builtin_amdgcn_wmma_f32_16x16x32_f16(false, a, false, tb, (short)0, c, false, false);
  }
  int ncol = o_base + (lane & 15);
  int mrow = m_base + ((lane >> 4) << 3);
  float bias = b[ncol];
  #pragma unroll
  for (int r = 0; r < 8; ++r)
    Y[(size_t)(mrow + r) * CIN + ncol] = c[r] + bias;
}

// ---------- launch ----------

extern "C" void kernel_launch(void* const* d_in, const int* in_sizes, int n_in,
                              void* d_out, int out_size, void* d_ws, size_t ws_size,
                              hipStream_t stream) {
  const float* x      = (const float*)d_in[0];
  const float* kpts   = (const float*)d_in[1];
  const float* pts3d  = (const float*)d_in[2];
  const float* pe_w1  = (const float*)d_in[3];
  const float* pe_b1  = (const float*)d_in[4];
  const float* pe_g1  = (const float*)d_in[5];
  const float* pe_bn1 = (const float*)d_in[6];
  const float* pe_w2  = (const float*)d_in[7];
  const float* pe_b2  = (const float*)d_in[8];
  const float* pe_g2  = (const float*)d_in[9];
  const float* pe_bn2 = (const float*)d_in[10];
  const float* lin_l_w = (const float*)d_in[11];
  const float* lin_l_b = (const float*)d_in[12];
  const float* lin_r_w = (const float*)d_in[13];
  const float* lin_r_b = (const float*)d_in[14];
  const float* lin_e_w = (const float*)d_in[15];
  const float* att     = (const float*)d_in[16];
  const float* conv_b  = (const float*)d_in[17];
  const float* norm_g  = (const float*)d_in[18];
  const float* norm_b  = (const float*)d_in[19];
  const float* res_w   = (const float*)d_in[20];
  const float* res_b   = (const float*)d_in[21];
  const float* proj_w  = (const float*)d_in[22];
  const float* proj_b  = (const float*)d_in[23];
  const int*   eidx    = (const int*)d_in[24];

  // workspace layout
  char* w = (char*)d_ws;
  auto alloc = [&](size_t bytes) -> char* {
    char* p = w;
    w += (bytes + 255) & ~(size_t)255;
    return p;
  };
  _Float16* hWl = (_Float16*)alloc((size_t)CIN * CIN * 2);
  _Float16* hWr = (_Float16*)alloc((size_t)CIN * CIN * 2);
  _Float16* hWs = (_Float16*)alloc((size_t)CIN * CIN * 2);
  _Float16* hWp = (_Float16*)alloc((size_t)CIN * CIN * 2);
  _Float16* xcomb = (_Float16*)alloc((size_t)N_NODES * CIN * 2);
  float* xl     = (float*)alloc((size_t)N_NODES * CIN * 4);
  float* xr     = (float*)alloc((size_t)N_NODES * CIN * 4);
  float* ident  = (float*)alloc((size_t)N_NODES * CIN * 4);
  float* outagg = (float*)alloc((size_t)N_NODES * CIN * 4);
  _Float16* act = (_Float16*)alloc((size_t)N_NODES * CIN * 2);
  float* logits = (float*)alloc((size_t)E_EDGES * HEADS * 4);
  unsigned* nmax = (unsigned*)alloc((size_t)N_NODES * HEADS * 4);
  float* nsum    = (float*)alloc((size_t)N_NODES * HEADS * 4);

  float* out_main  = (float*)d_out;                              // [N,256]
  float* out_alpha = out_main + (size_t)N_NODES * CIN;           // [E,4]
  float* out_ea    = out_alpha + (size_t)E_EDGES * HEADS;        // [E,3]

  // 1. convert weights
  k_cvt_weights<<<(CIN * CIN + 255) / 256, 256, 0, stream>>>(
      lin_l_w, lin_r_w, res_w, proj_w, hWl, hWr, hWs, hWp);
  // 2. init accumulators / softmax state
  k_init<<<(N_NODES * CIN + 255) / 256, 256, 0, stream>>>(outagg, nmax, nsum);
  // 3. positional encoder + concat
  k_posenc<<<(N_NODES + 3) / 4, 128, 0, stream>>>(
      x, kpts, pts3d, pe_w1, pe_b1, pe_g1, pe_bn1, pe_w2, pe_b2, pe_g2, pe_bn2, xcomb);
  // 4. triple WMMA GEMM (x_l, x_r, identity)
  dim3 g3(N_NODES / 16, CIN / 16 / 4);
  k_gemm3<<<g3, 128, 0, stream>>>(xcomb, hWl, hWr, hWs,
                                  lin_l_b, lin_r_b, res_b, xl, xr, ident);
  // 5-7. edge stages
  int eblocks = (E_EDGES * HEADS + 255) / 256;
  k_edge_logits<<<eblocks, 256, 0, stream>>>(eidx, kpts, xl, xr, lin_e_w, att,
                                             logits, nmax, out_ea);
  k_edge_exp<<<eblocks, 256, 0, stream>>>(eidx, logits, nmax, nsum);
  k_edge_scatter<<<eblocks, 256, 0, stream>>>(eidx, logits, nsum, xl, outagg, out_alpha);
  // 8. LN + residual + SiLU
  k_epilogue<<<(N_NODES + 3) / 4, 128, 0, stream>>>(outagg, conv_b, norm_g, norm_b,
                                                    ident, act);
  // 9. projection WMMA GEMM -> final output
  k_proj<<<g3, 128, 0, stream>>>(act, hWp, proj_b, out_main);
}
